// Mamba_79766132621892
// MI455X (gfx1250) — matrix operand order repaired and checked by reading, not compiled, not run on hardware
//
#include <hip/hip_runtime.h>
#include <hip/hip_bf16.h>
#include <math.h>

// ---------------- problem constants ----------------
constexpr int BATCH   = 4;
constexpr int LSEQ    = 2048;
constexpr int ROWS    = BATCH * LSEQ;          // 8192
constexpr int DMODEL  = 512;
constexpr int DINNER  = 1024;
constexpr int DSTATE  = 128;
constexpr int HEADDIM = 64;
constexpr int NH      = 16;
constexpr int DCONV   = 4;
constexpr int CONVD   = DINNER + 2 * DSTATE;   // 1280
constexpr int DPROJ   = 2 * DINNER + 2 * DSTATE + NH; // 2320
constexpr int NPAD1   = 2432;                  // 19 * 128, padded N for in_proj weight
constexpr float EPS   = 1e-5f;

// Async direct-to-LDS path (gfx1250): guarded, falls back to sync staging.
#if defined(__has_builtin)
#  if __has_builtin(__builtin_amdgcn_global_load_async_to_lds_b128)
#    define USE_ASYNC_LDS 1
#  endif
#endif
#ifndef USE_ASYNC_LDS
#  define USE_ASYNC_LDS 0
#endif

// Exact parameter type reported by the compiler for the async builtin.
typedef int v4i_async __attribute__((vector_size(4 * sizeof(int))));

// WMMA fragment types (CDNA5 gfx1250, wave32)
typedef __attribute__((ext_vector_type(16))) __bf16 v16bf;
typedef __attribute__((ext_vector_type(8)))  float  v8f;

union FragAB { v16bf v; unsigned int u[8]; };

__device__ __forceinline__ float silu_f(float x) {
    return x / (1.f + __expf(-x));
}

// ---------------- tiny conversion kernels ----------------
__global__ void f32_to_bf16_kernel(const float* __restrict__ src,
                                   __bf16* __restrict__ dst, int n) {
    int i = blockIdx.x * blockDim.x + threadIdx.x;
    if (i < n) dst[i] = (__bf16)src[i];
}

// src: [K][N] fp32 row-major  ->  dst: [Npad][K] bf16 (zero padded for n>=N)
__global__ void transpose_w_kernel(const float* __restrict__ src,
                                   __bf16* __restrict__ dst,
                                   int K, int N, int Npad) {
    int idx = blockIdx.x * blockDim.x + threadIdx.x;
    if (idx >= Npad * K) return;
    int n = idx / K;
    int k = idx - n * K;
    dst[idx] = (n < N) ? (__bf16)src[(size_t)k * N + n] : (__bf16)0.f;
}

// ---------------- bf16 WMMA GEMM: C[M][ldc] = A[M][K] * Bt[N][K]^T ----------------
// BM=128 BN=128 BK=32, 256 threads = 8 waves (2x4), each wave 64x32 = 8 WMMA tiles.
__global__ __launch_bounds__(256)
void gemm_bf16_wmma_kernel(const __bf16* __restrict__ A,
                           const __bf16* __restrict__ Bt,
                           float* __restrict__ C,
                           int K, int Nvalid, int ldc) {
    constexpr int BM = 128, BN = 128, BK = 32;
    __shared__ __align__(16) __bf16 sA[BM * BK];
    __shared__ __align__(16) __bf16 sB[BN * BK];

    const int t    = threadIdx.x;
    const int lane = t & 31;
    const int wv   = t >> 5;      // 0..7
    const int wm   = wv >> 2;     // 0..1 : 64-row band
    const int wn   = wv & 3;      // 0..3 : 32-col band
    const int lrow = lane & 15;
    const int lhi  = lane >> 4;

    const int m0 = blockIdx.y * BM;
    const int n0 = blockIdx.x * BN;
    const int ktiles = K / BK;

    v8f acc[4][2] = {};

    for (int kt = 0; kt < ktiles; ++kt) {
        // ---- stage A/B tiles into LDS (32B per thread per matrix) ----
        #pragma unroll
        for (int it = 0; it < 2; ++it) {
            int i = t + it * 256;          // 0..511
            int r = i >> 2;
            int c = i & 3;
            const __bf16* gA = A  + (size_t)(m0 + r) * K + (size_t)kt * BK + c * 8;
            const __bf16* gB = Bt + (size_t)(n0 + r) * K + (size_t)kt * BK + c * 8;
#if USE_ASYNC_LDS
            __builtin_amdgcn_global_load_async_to_lds_b128(
                (v4i_async*)gA, (v4i_async*)&sA[i * 8], 0, 0);
            __builtin_amdgcn_global_load_async_to_lds_b128(
                (v4i_async*)gB, (v4i_async*)&sB[i * 8], 0, 0);
#else
            ((uint4*)sA)[i] = *(const uint4*)gA;
            ((uint4*)sB)[i] = *(const uint4*)gB;
#endif
        }
        if (kt + 1 < ktiles) {
            __builtin_prefetch(A + (size_t)(m0 + (t >> 2)) * K + (size_t)(kt + 1) * BK, 0, 0);
            __builtin_prefetch(Bt + (size_t)(n0 + (t >> 2)) * K + (size_t)(kt + 1) * BK, 0, 0);
        }
#if USE_ASYNC_LDS
#  if __has_builtin(__builtin_amdgcn_s_wait_asynccnt)
        __builtin_amdgcn_s_wait_asynccnt(0);
#  else
        asm volatile("s_wait_asynccnt 0x0" ::: "memory");
#  endif
#endif
        __syncthreads();

        // ---- build fragments from LDS per ISA layouts ----
        const unsigned int* uA = (const unsigned int*)sA;
        const unsigned int* uB = (const unsigned int*)sB;

        FragAB af[4];
        #pragma unroll
        for (int mt = 0; mt < 4; ++mt) {
            int row = wm * 64 + mt * 16 + lrow;
            #pragma unroll
            for (int v = 0; v < 8; ++v)
                af[mt].u[v] = uA[row * 16 + lhi * 4 + (v & 3) + ((v >> 2) << 3)];
        }
        FragAB bfr[2];
        #pragma unroll
        for (int nt = 0; nt < 2; ++nt) {
            int col = wn * 32 + nt * 16 + lrow;
            #pragma unroll
            for (int v = 0; v < 8; ++v)
                bfr[nt].u[v] = uB[col * 16 + lhi * 4 + (v & 3) + ((v >> 2) << 3)];
        }

        #pragma unroll
        for (int mt = 0; mt < 4; ++mt)
            #pragma unroll
            for (int nt = 0; nt < 2; ++nt)
                acc[mt][nt] = __builtin_amdgcn_wmma_f32_16x16x32_bf16(
                    false, af[mt].v, false, bfr[nt].v,
                    (short)0, acc[mt][nt], false, false);

        __syncthreads();
    }

    // ---- store (C/D layout: lane<16 -> M=r, lane>=16 -> M=r+8; N = lane&15) ----
    #pragma unroll
    for (int mt = 0; mt < 4; ++mt) {
        #pragma unroll
        for (int nt = 0; nt < 2; ++nt) {
            int gm = m0 + wm * 64 + mt * 16 + lhi * 8;
            int gn = n0 + wn * 32 + nt * 16 + lrow;
            if (gn < Nvalid) {
                #pragma unroll
                for (int r = 0; r < 8; ++r)
                    C[(size_t)(gm + r) * ldc + gn] = acc[mt][nt][r];
            }
        }
    }
}

// ---------------- depthwise causal conv (width 4) + SiLU ----------------
__global__ void conv_silu_kernel(const float* __restrict__ zx,
                                 const float* __restrict__ conv_w,   // [4][CONVD]
                                 const float* __restrict__ conv_b,   // [CONVD]
                                 float* __restrict__ xbc) {
    int rc = blockIdx.x;              // 0..ROWS-1
    int b  = rc / LSEQ;
    int l  = rc - b * LSEQ;
    for (int c = threadIdx.x; c < CONVD; c += blockDim.x) {
        float acc = conv_b[c];
        #pragma unroll
        for (int k = 0; k < DCONV; ++k) {
            int ls = l - (DCONV - 1) + k;
            if (ls >= 0)
                acc += conv_w[k * CONVD + c] *
                       zx[(size_t)(b * LSEQ + ls) * DPROJ + DINNER + c];
        }
        xbc[(size_t)rc * CONVD + c] = silu_f(acc);
    }
}

// ---------------- dt = softplus(dt_raw + bias), dA = exp(-exp(A_log)*dt) ----------------
__global__ void dt_kernel(const float* __restrict__ zx,
                          const float* __restrict__ dt_bias,
                          const float* __restrict__ A_log,
                          float* __restrict__ dt_arr,
                          float* __restrict__ dA_arr) {
    int idx = blockIdx.x * blockDim.x + threadIdx.x; // ROWS*NH
    int r = idx >> 4;
    int h = idx & 15;
    float v  = zx[(size_t)r * DPROJ + (2 * DINNER + 2 * DSTATE) + h] + dt_bias[h];
    float sp = (v > 20.f) ? v : log1pf(__expf(v));
    dt_arr[idx] = sp;
    dA_arr[idx] = __expf(-__expf(A_log[h]) * sp);
}

// ---------------- selective scan: one block per (batch, head) ----------------
// 256 threads: thread t owns state h[p][n0..n0+31], p = t>>2, n0 = (t&3)*32.
__global__ __launch_bounds__(256)
void scan_kernel(const float* __restrict__ xbc,
                 const float* __restrict__ dt_arr,
                 const float* __restrict__ dA_arr,
                 const float* __restrict__ D_skip,
                 float* __restrict__ y) {
    int bh = blockIdx.x;
    int b  = bh >> 4;
    int h  = bh & 15;
    int t  = threadIdx.x;
    int p  = t >> 2;
    int nq = (t & 3) * 32;

    __shared__ float sB[DSTATE], sC[DSTATE], sX[HEADDIM], sScal[2];

    float s[32];
    #pragma unroll
    for (int i = 0; i < 32; ++i) s[i] = 0.f;
    const float dsk = D_skip[h];

    for (int step = 0; step < LSEQ; ++step) {
        int row = b * LSEQ + step;
        if (t < DSTATE)
            sB[t] = xbc[(size_t)row * CONVD + DINNER + t];
        else
            sC[t - DSTATE] = xbc[(size_t)row * CONVD + DINNER + DSTATE + (t - DSTATE)];
        if (t < HEADDIM)
            sX[t] = xbc[(size_t)row * CONVD + h * HEADDIM + t];
        if (t == 0) {
            sScal[0] = dt_arr[row * NH + h];
            sScal[1] = dA_arr[row * NH + h];
        }
        __syncthreads();

        float dtv = sScal[0], dAv = sScal[1];
        float xp  = sX[p];
        float coef = dtv * xp;
        float part = 0.f;
        #pragma unroll
        for (int i = 0; i < 32; ++i) {
            s[i] = dAv * s[i] + coef * sB[nq + i];
            part += s[i] * sC[nq + i];
        }
        part += __shfl_xor(part, 1, 32);
        part += __shfl_xor(part, 2, 32);
        if ((t & 3) == 0)
            y[(size_t)row * DINNER + h * HEADDIM + p] = part + dsk * xp;
        __syncthreads();
    }
}

// ---------------- y * silu(z), RMSNorm, * norm_w -> bf16 ----------------
__global__ __launch_bounds__(256)
void gated_norm_kernel(const float* __restrict__ y,
                       const float* __restrict__ zx,
                       const float* __restrict__ norm_w,
                       __bf16* __restrict__ out_bf) {
    int r = blockIdx.x;
    __shared__ float wsum[8];
    __shared__ float stot;
    float vals[4];
    float loc = 0.f;
    #pragma unroll
    for (int j = 0; j < 4; ++j) {
        int c = threadIdx.x + j * 256;
        float z = zx[(size_t)r * DPROJ + c];
        float v = y[(size_t)r * DINNER + c] * silu_f(z);
        vals[j] = v;
        loc += v * v;
    }
    #pragma unroll
    for (int o = 16; o > 0; o >>= 1) loc += __shfl_xor(loc, o, 32);
    if ((threadIdx.x & 31) == 0) wsum[threadIdx.x >> 5] = loc;
    __syncthreads();
    if (threadIdx.x == 0) {
        float ssum = 0.f;
        #pragma unroll
        for (int i = 0; i < 8; ++i) ssum += wsum[i];
        stot = rsqrtf(ssum / (float)DINNER + EPS);
    }
    __syncthreads();
    float rr = stot;
    #pragma unroll
    for (int j = 0; j < 4; ++j) {
        int c = threadIdx.x + j * 256;
        out_bf[(size_t)r * DINNER + c] = (__bf16)(vals[j] * rr * norm_w[c]);
    }
}

// ---------------- workspace layout (bytes) ----------------
constexpr size_t SZ_ABF   = (size_t)ROWS * DINNER * 2;       // bf16 activations (x or y_norm)
constexpr size_t SZ_WBIN  = (size_t)NPAD1 * DMODEL * 2;      // in_w  bf16, [Npad][K]
constexpr size_t SZ_WBOUT = (size_t)DMODEL * DINNER * 2;     // out_w bf16, [N][K]
constexpr size_t SZ_ZX    = (size_t)ROWS * DPROJ * 4;
constexpr size_t SZ_XBC   = (size_t)ROWS * CONVD * 4;
constexpr size_t SZ_DT    = (size_t)ROWS * NH * 4;
constexpr size_t SZ_Y     = (size_t)ROWS * DINNER * 4;

constexpr size_t OFF_ABF   = 0;
constexpr size_t OFF_WBIN  = OFF_ABF   + SZ_ABF;
constexpr size_t OFF_WBOUT = OFF_WBIN  + SZ_WBIN;
constexpr size_t OFF_ZX    = OFF_WBOUT + SZ_WBOUT;
constexpr size_t OFF_XBC   = OFF_ZX    + SZ_ZX;
constexpr size_t OFF_DT    = OFF_XBC   + SZ_XBC;
constexpr size_t OFF_DA    = OFF_DT    + SZ_DT;
constexpr size_t OFF_Y     = OFF_DA    + SZ_DT;
constexpr size_t OFF_XN    = OFF_Y     + SZ_Y;

extern "C" void kernel_launch(void* const* d_in, const int* in_sizes, int n_in,
                              void* d_out, int out_size, void* d_ws, size_t ws_size,
                              hipStream_t stream) {
    (void)in_sizes; (void)n_in; (void)out_size; (void)ws_size;

    const float* x_in    = (const float*)d_in[0];
    const float* in_w    = (const float*)d_in[1];
    const float* conv_w  = (const float*)d_in[2];
    const float* conv_b  = (const float*)d_in[3];
    const float* dt_bias = (const float*)d_in[4];
    const float* A_log   = (const float*)d_in[5];
    const float* D_skip  = (const float*)d_in[6];
    const float* norm_w  = (const float*)d_in[7];
    const float* out_w   = (const float*)d_in[8];

    char* ws = (char*)d_ws;
    __bf16* abf   = (__bf16*)(ws + OFF_ABF);
    __bf16* wbin  = (__bf16*)(ws + OFF_WBIN);
    __bf16* wbout = (__bf16*)(ws + OFF_WBOUT);
    float*  zx    = (float*)(ws + OFF_ZX);
    float*  xbc   = (float*)(ws + OFF_XBC);
    float*  dtv   = (float*)(ws + OFF_DT);
    float*  dav   = (float*)(ws + OFF_DA);
    float*  ybuf  = (float*)(ws + OFF_Y);
    float*  xnext = (float*)(ws + OFF_XN);

    const float* cur = x_in;
    for (int layer = 0; layer < 2; ++layer) {
        const float* in_w_l    = in_w    + (size_t)layer * DMODEL * DPROJ;
        const float* conv_w_l  = conv_w  + (size_t)layer * DCONV * CONVD;
        const float* conv_b_l  = conv_b  + (size_t)layer * CONVD;
        const float* dt_bias_l = dt_bias + (size_t)layer * NH;
        const float* A_log_l   = A_log   + (size_t)layer * NH;
        const float* D_skip_l  = D_skip  + (size_t)layer * NH;
        const float* norm_w_l  = norm_w  + (size_t)layer * DINNER;
        const float* out_w_l   = out_w   + (size_t)layer * DINNER * DMODEL;

        // 1) x -> bf16
        f32_to_bf16_kernel<<<(ROWS * DMODEL + 255) / 256, 256, 0, stream>>>(
            cur, abf, ROWS * DMODEL);
        // 2) in_w -> bf16 [Npad][K], zero padded
        transpose_w_kernel<<<(NPAD1 * DMODEL + 255) / 256, 256, 0, stream>>>(
            in_w_l, wbin, DMODEL, DPROJ, NPAD1);
        // 3) zxbcdt = x @ in_w   (WMMA)
        gemm_bf16_wmma_kernel<<<dim3(NPAD1 / 128, ROWS / 128), 256, 0, stream>>>(
            abf, wbin, zx, DMODEL, DPROJ, DPROJ);
        // 4) dt / dA
        dt_kernel<<<(ROWS * NH) / 256, 256, 0, stream>>>(
            zx, dt_bias_l, A_log_l, dtv, dav);
        // 5) conv + silu
        conv_silu_kernel<<<ROWS, 256, 0, stream>>>(zx, conv_w_l, conv_b_l, xbc);
        // 6) selective scan
        scan_kernel<<<BATCH * NH, 256, 0, stream>>>(xbc, dtv, dav, D_skip_l, ybuf);
        // 7) gated RMSNorm -> bf16
        gated_norm_kernel<<<ROWS, 256, 0, stream>>>(ybuf, zx, norm_w_l, abf);
        // 8) out_w -> bf16 [N][K]
        transpose_w_kernel<<<(DMODEL * DINNER + 255) / 256, 256, 0, stream>>>(
            out_w_l, wbout, DINNER, DMODEL, DMODEL);
        // 9) x_next = y_norm @ out_w   (WMMA)
        float* dst = (layer == 1) ? (float*)d_out : xnext;
        gemm_bf16_wmma_kernel<<<dim3(DMODEL / 128, ROWS / 128), 256, 0, stream>>>(
            abf, wbout, dst, DINNER, DMODEL, DMODEL);

        cur = xnext;
    }
}
